// Attention_56633438765665
// MI455X (gfx1250) — matrix operand order repaired and checked by reading, not compiled, not run on hardware
//
#include <hip/hip_runtime.h>
#include <cstdint>
#include <cstddef>

typedef unsigned short u16;
typedef __bf16 bf16_t;
typedef bf16_t v16bf __attribute__((ext_vector_type(16)));
typedef float  v8f   __attribute__((ext_vector_type(8)));

union Frag {
  v16bf bf;
  uint4 q[2];
  u16   u[16];
};

__device__ __forceinline__ u16 f2bf(float f) {
  unsigned int u = __float_as_uint(f);
  u += 0x7FFFu + ((u >> 16) & 1u);   // round-to-nearest-even
  return (u16)(u >> 16);
}
__device__ __forceinline__ float bf2f(u16 h) {
  return __uint_as_float(((unsigned int)h) << 16);
}

__device__ __forceinline__ v8f wmma_bf16(const Frag& a, const Frag& b, v8f c) {
  // D = A(16x32 bf16) * B(32x16 bf16) + C(16x16 f32)
  return __builtin_amdgcn_wmma_f32_16x16x32_bf16(false, a.bf, false, b.bf,
                                                 (short)0, c, false, false);
}

// ---- CDNA5 async global->LDS copy (ASYNCcnt path), layout-preserving 16B ----
__device__ __forceinline__ void async_load_b128(unsigned lds_byte_off,
                                                const void* gaddr) {
  asm volatile("global_load_async_to_lds_b128 %0, %1, off"
               :: "v"(lds_byte_off), "v"(gaddr)
               : "memory");
}
__device__ __forceinline__ void wait_asynccnt0() {
  asm volatile("s_wait_asynccnt 0x0" ::: "memory");
}
__device__ __forceinline__ unsigned lds_off(const void* p) {
  return (unsigned)(unsigned long long)p;  // low 32 bits = LDS aperture offset
}

// reductions across the 16-lane group that holds one C/D row
__device__ __forceinline__ float red_max16(float v) {
  v = fmaxf(v, __shfl_xor(v, 1, 32));
  v = fmaxf(v, __shfl_xor(v, 2, 32));
  v = fmaxf(v, __shfl_xor(v, 4, 32));
  v = fmaxf(v, __shfl_xor(v, 8, 32));
  return v;
}
__device__ __forceinline__ float red_sum16(float v) {
  v += __shfl_xor(v, 1, 32);
  v += __shfl_xor(v, 2, 32);
  v += __shfl_xor(v, 4, 32);
  v += __shfl_xor(v, 8, 32);
  return v;
}

// ---------------------------------------------------------------------------
// f32 -> bf16 conversion (weights)
// ---------------------------------------------------------------------------
__global__ __launch_bounds__(256) void cvt_f32_bf16(const float* __restrict__ in,
                                                    u16* __restrict__ out, int n) {
  int i = (blockIdx.x * 256 + threadIdx.x) * 4;
  if (i >= n) return;
  float4 v = *(const float4*)(in + i);
  out[i + 0] = f2bf(v.x);
  out[i + 1] = f2bf(v.y);
  out[i + 2] = f2bf(v.z);
  out[i + 3] = f2bf(v.w);
}

// ---------------------------------------------------------------------------
// Tiled bf16 WMMA GEMM: C[M x NC] = A[M x K] * B[K x NC]
// Block: 256 threads (8 waves), 128x128 output tile, BK = 32.
// Each wave: 32x64 sub-tile = 2x4 grid of 16x16 WMMA accumulators.
// out_mode 0: write bf16 scattered into [B,H,N,hd] layout (split K/V buffers)
// out_mode 1: write f32 row-major + bias (final projection)
// ---------------------------------------------------------------------------
#define LDSA_STRIDE 40   // halfwords; 80B rows -> 16B aligned frag chunks
#define LDSB_STRIDE 40

__global__ __launch_bounds__(256) void gemm_bf16_kernel(
    const void* __restrict__ Av, int a_is_bf16,
    const u16* __restrict__ Bw,
    int K, int NC,
    int out_mode,
    u16* __restrict__ dstA, u16* __restrict__ dstB, int split,
    float* __restrict__ outF, const float* __restrict__ bias,
    int Nseq, int Hh) {
  __shared__ __attribute__((aligned(16))) u16 lds_a[128 * LDSA_STRIDE];
  __shared__ __attribute__((aligned(16))) u16 lds_b[128 * LDSB_STRIDE];

  const int t    = threadIdx.x;
  const int lane = t & 31, w = t >> 5;
  const int m    = lane & 15, h2 = lane >> 4;
  const int wm   = (w >> 1) * 32, wn = (w & 1) * 64;
  const int rowBase = blockIdx.y * 128;
  const int colBase = blockIdx.x * 128;

  const int ar  = t >> 1;          // staging: A row 0..127
  const int akg = (t & 1) * 16;    //          A k-group 0/16
  const int bk  = t >> 3;          //          B k 0..31
  const int bcg = (t & 7) * 16;    //          B col-group

  v8f acc[2][4] = {};

  const int ksteps = K / 32;
  for (int kt = 0; kt < ksteps; ++kt) {
    const int k0 = kt * 32;
    __syncthreads();

    // ---- stage A tile (128x32) row-major bf16 into LDS ----
    if (a_is_bf16) {
      const u16* A = (const u16*)Av;
      const u16* p = A + (size_t)(rowBase + ar) * K + k0 + akg;
      const unsigned dst = lds_off(&lds_a[ar * LDSA_STRIDE + akg]);
      async_load_b128(dst, p);          // ASYNCcnt DMA, no VGPR round-trip
      async_load_b128(dst + 16, p + 8);
    } else {
      const float* A = (const float*)Av;
      const float* p = A + (size_t)(rowBase + ar) * K + k0 + akg;
      float4 f0 = *(const float4*)(p + 0);
      float4 f1 = *(const float4*)(p + 4);
      float4 f2 = *(const float4*)(p + 8);
      float4 f3 = *(const float4*)(p + 12);
      union { uint4 q[2]; u16 u[16]; } pk;
      pk.u[0]  = f2bf(f0.x); pk.u[1]  = f2bf(f0.y); pk.u[2]  = f2bf(f0.z); pk.u[3]  = f2bf(f0.w);
      pk.u[4]  = f2bf(f1.x); pk.u[5]  = f2bf(f1.y); pk.u[6]  = f2bf(f1.z); pk.u[7]  = f2bf(f1.w);
      pk.u[8]  = f2bf(f2.x); pk.u[9]  = f2bf(f2.y); pk.u[10] = f2bf(f2.z); pk.u[11] = f2bf(f2.w);
      pk.u[12] = f2bf(f3.x); pk.u[13] = f2bf(f3.y); pk.u[14] = f2bf(f3.z); pk.u[15] = f2bf(f3.w);
      *(uint4*)&lds_a[ar * LDSA_STRIDE + akg]     = pk.q[0];
      *(uint4*)&lds_a[ar * LDSA_STRIDE + akg + 8] = pk.q[1];
      if (kt + 1 < ksteps)
        __builtin_prefetch(p + 32, 0, 1);   // next k-step A line -> global_prefetch
    }

    // ---- stage B tile (32x128) transposed: lds_b[col][k] ----
    {
      const u16* p = Bw + (size_t)(k0 + bk) * NC + colBase + bcg;
      union { uint4 q; u16 u[8]; } u0, u1;
      u0.q = *(const uint4*)p;
      u1.q = *(const uint4*)(p + 8);
#pragma unroll
      for (int i = 0; i < 8; ++i) lds_b[(bcg + i) * LDSB_STRIDE + bk] = u0.u[i];
#pragma unroll
      for (int i = 0; i < 8; ++i) lds_b[(bcg + 8 + i) * LDSB_STRIDE + bk] = u1.u[i];
      if (kt + 1 < ksteps)
        __builtin_prefetch(p + (size_t)32 * NC, 0, 1);  // next k-step B line
    }
    wait_asynccnt0();
    __syncthreads();

    // ---- fragment loads per ISA layouts ----
    Frag a[2], bfr[4];
#pragma unroll
    for (int mt = 0; mt < 2; ++mt) {
      // A elem e<8: k = h2*8+e ; e>=8: k = 16+h2*8+(e-8)
      const u16* p = &lds_a[(wm + mt * 16 + m) * LDSA_STRIDE + h2 * 8];
      a[mt].q[0] = *(const uint4*)p;
      a[mt].q[1] = *(const uint4*)(p + 16);
    }
#pragma unroll
    for (int nt = 0; nt < 4; ++nt) {
      // B elem e<8: k = h2*8+e ; e>=8: k = 16+h2*8+(e-8), col = lane&15
      const u16* p = &lds_b[(wn + nt * 16 + m) * LDSB_STRIDE + h2 * 8];
      bfr[nt].q[0] = *(const uint4*)p;
      bfr[nt].q[1] = *(const uint4*)(p + 16);
    }
#pragma unroll
    for (int mt = 0; mt < 2; ++mt)
#pragma unroll
      for (int nt = 0; nt < 4; ++nt)
        acc[mt][nt] = wmma_bf16(a[mt], bfr[nt], acc[mt][nt]);
  }

  // ---- epilogue: C elem r -> row h2*8+r, col = lane&15 ----
  if (out_mode == 1) {
#pragma unroll
    for (int mt = 0; mt < 2; ++mt)
#pragma unroll
      for (int nt = 0; nt < 4; ++nt)
#pragma unroll
        for (int r = 0; r < 8; ++r) {
          const int grow = rowBase + wm + mt * 16 + h2 * 8 + r;
          const int gcol = colBase + wn + nt * 16 + m;
          outF[(size_t)grow * NC + gcol] = acc[mt][nt][r] + bias[gcol];
        }
  } else {
    // whole 128-row block lies in one batch (Nseq % 128 == 0): hoist div/mod
    const int bidx  = rowBase / Nseq;
    const int nbase = rowBase % Nseq;
#pragma unroll
    for (int mt = 0; mt < 2; ++mt) {
#pragma unroll
      for (int nt = 0; nt < 4; ++nt) {
        const int gcol = colBase + wn + nt * 16 + m;
        int cc = gcol;
        u16* base = dstA;
        if (cc >= split) { cc -= split; base = dstB; }
        const int hh = cc >> 6, d = cc & 63;   // hd = 64
        u16* bp = base + (((size_t)(bidx * Hh + hh)) * Nseq) * 64 + d;
#pragma unroll
        for (int r = 0; r < 8; ++r) {
          const int n = nbase + wm + mt * 16 + h2 * 8 + r;
          bp[(size_t)n * 64] = f2bf(acc[mt][nt][r]);
        }
      }
    }
  }
}

// ---------------------------------------------------------------------------
// In-place rotary embedding on bf16 [B,H,N,hd] buffer (hd=64).
// Each thread owns the (d, d+32) pair -> in-place safe.
// Score scale (hd^-0.5) is folded in via `scale` for Q.
// ---------------------------------------------------------------------------
__global__ __launch_bounds__(256) void rotary_kernel(u16* __restrict__ buf,
                                                     const float* __restrict__ pos,
                                                     int N, float scale, int total) {
  int idx = blockIdx.x * 256 + threadIdx.x;
  if (idx >= total) return;
  int d  = idx & 31;
  int n  = (idx >> 5) % N;
  int bh = idx / (N * 32);
  u16* p = buf + ((size_t)bh * N + n) * 64;
  float q1 = bf2f(p[d]), q2 = bf2f(p[d + 32]);
  float p1 = pos[n * 64 + d], p2 = pos[n * 64 + d + 32];
  float o1 = q1 * __cosf(p1) - q2 * __sinf(p1);   // first half: -t2*sin
  float o2 = q2 * __cosf(p2) + q1 * __sinf(p2);   // second half: +t1*sin
  p[d]      = f2bf(o1 * scale);
  p[d + 32] = f2bf(o2 * scale);
}

// ---------------------------------------------------------------------------
// Flash attention. Q/K/V: bf16 [B,H,N,64] (scale folded into Q).
// Grid: (N/128, B*H); 256 threads = 8 waves; wave owns 16 query rows.
// Per 32-key chunk: K staged via async-to-LDS DMA, V transposed via VGPRs,
// 4 WMMAs for scores, online softmax with shuffle reductions,
// LDS transpose of P, 4 WMMAs for PV.  Output: bf16 [B*N, D].
// ---------------------------------------------------------------------------
__global__ __launch_bounds__(256) void flash_attn_kernel(
    const u16* __restrict__ Q, const u16* __restrict__ Kc,
    const u16* __restrict__ V, u16* __restrict__ Aout,
    int Bt, int Hh, int N) {
  (void)Bt;
  __shared__ __attribute__((aligned(16))) u16 lds_k[32 * 72];      // [key][dim]
  __shared__ __attribute__((aligned(16))) u16 lds_vt[64 * 40];     // [dim][key]
  __shared__ __attribute__((aligned(16))) u16 lds_p[8 * 16 * 40];  // per-wave P

  const int t    = threadIdx.x;
  const int lane = t & 31, w = t >> 5;
  const int m    = lane & 15, h2 = lane >> 4;
  const int bh   = blockIdx.y;
  const size_t hbase = (size_t)bh * N * 64;
  const u16* Qb = Q + hbase;
  const u16* Kb = Kc + hbase;
  const u16* Vb = V + hbase;

  // Q fragments: 16 rows x 64 dims = two 16x32 A-frags, loaded once
  const int qrow = blockIdx.x * 128 + w * 16 + m;
  Frag aq[2];
#pragma unroll
  for (int j = 0; j < 2; ++j) {
    const u16* p = Qb + (size_t)qrow * 64 + j * 32 + h2 * 8;
    aq[j].q[0] = *(const uint4*)p;
    aq[j].q[1] = *(const uint4*)(p + 16);
  }

  float mrun[8], lrun[8];
#pragma unroll
  for (int r = 0; r < 8; ++r) { mrun[r] = -1e30f; lrun[r] = 0.0f; }
  v8f o[4] = {};

  const int skey = t >> 3;        // staging: key 0..31
  const int sdg  = (t & 7) * 8;   //          dim group
  u16* pw = &lds_p[w * 16 * 40];
  const unsigned kdst = lds_off(&lds_k[skey * 72 + sdg]);

  for (int kc = 0; kc < N / 32; ++kc) {
    __syncthreads();
    {
      const size_t kro = (size_t)(kc * 32 + skey) * 64 + sdg;
      // K tile: one-instruction async DMA into LDS (layout preserving)
      async_load_b128(kdst, Kb + kro);
      // V tile: needs transpose -> VGPR path
      union { uint4 q; u16 u[8]; } vv;
      vv.q = *(const uint4*)(Vb + kro);
#pragma unroll
      for (int i = 0; i < 8; ++i) lds_vt[(sdg + i) * 40 + skey] = vv.u[i];
    }
    wait_asynccnt0();
    __syncthreads();

#pragma unroll
    for (int sub = 0; sub < 2; ++sub) {
      Frag bk0, bk1;  // B-frags: k = hd dim, n = key
      {
        const u16* p = &lds_k[(sub * 16 + m) * 72 + h2 * 8];
        bk0.q[0] = *(const uint4*)p;        bk0.q[1] = *(const uint4*)(p + 16);
        bk1.q[0] = *(const uint4*)(p + 32); bk1.q[1] = *(const uint4*)(p + 48);
      }
      v8f s = {};
      s = wmma_bf16(aq[0], bk0, s);
      s = wmma_bf16(aq[1], bk1, s);

#pragma unroll
      for (int r = 0; r < 8; ++r) {
        float sv    = s[r];
        float mx    = red_max16(sv);
        float mnew  = fmaxf(mrun[r], mx);
        float alpha = __expf(mrun[r] - mnew);
        float pv    = __expf(sv - mnew);
        float rs    = red_sum16(pv);
        lrun[r] = lrun[r] * alpha + rs;
        mrun[r] = mnew;
#pragma unroll
        for (int nt = 0; nt < 4; ++nt) o[nt][r] *= alpha;
        // C-layout -> LDS [row][key] for A-layout reload
        pw[(h2 * 8 + r) * 40 + sub * 16 + m] = f2bf(pv);
      }
    }

    // P as 16x32 A-frag (transposed via per-wave LDS scratch)
    Frag ap;
    {
      const u16* p = pw + m * 40 + h2 * 8;
      ap.q[0] = *(const uint4*)p;
      ap.q[1] = *(const uint4*)(p + 16);
    }
#pragma unroll
    for (int nt = 0; nt < 4; ++nt) {
      Frag bv;  // k = key, n = dim
      const u16* p = &lds_vt[(nt * 16 + m) * 40 + h2 * 8];
      bv.q[0] = *(const uint4*)p;
      bv.q[1] = *(const uint4*)(p + 16);
      o[nt] = wmma_bf16(ap, bv, o[nt]);
    }
  }

  const int b = bh / Hh, hh = bh % Hh;
#pragma unroll
  for (int r = 0; r < 8; ++r) {
    const float inv = 1.0f / lrun[r];
    const int row = blockIdx.x * 128 + w * 16 + h2 * 8 + r;
    u16* op = Aout + ((size_t)(b * N + row)) * 1024 + hh * 64;
#pragma unroll
    for (int nt = 0; nt < 4; ++nt)
      op[nt * 16 + m] = f2bf(o[nt][r] * inv);
  }
}

// ---------------------------------------------------------------------------
extern "C" void kernel_launch(void* const* d_in, const int* in_sizes, int n_in,
                              void* d_out, int out_size, void* d_ws, size_t ws_size,
                              hipStream_t stream) {
  (void)in_sizes; (void)n_in; (void)out_size; (void)ws_size;
  const float* x    = (const float*)d_in[0];
  const float* ctx  = (const float*)d_in[1];
  const float* pos  = (const float*)d_in[2];
  const float* Wq   = (const float*)d_in[3];
  const float* Wkv  = (const float*)d_in[4];
  const float* Wout = (const float*)d_in[5];
  const float* bout = (const float*)d_in[6];
  float* out = (float*)d_out;

  const int B = 2, N = 2048, D = 1024, H = 16;
  const int M = B * N;

  char* ws = (char*)d_ws;
  size_t off = 0;
  auto wsalloc = [&](size_t bytes) -> void* {
    void* p = ws + off;
    off += (bytes + 255) & ~(size_t)255;
    return p;
  };
  u16* Wq_bf   = (u16*)wsalloc((size_t)D * D * 2);
  u16* Wkv_bf  = (u16*)wsalloc((size_t)D * 2 * D * 2);
  u16* Wout_bf = (u16*)wsalloc((size_t)D * D * 2);
  u16* Qbf     = (u16*)wsalloc((size_t)M * D * 2);
  u16* Kbf     = (u16*)wsalloc((size_t)M * D * 2);
  u16* Vbf     = (u16*)wsalloc((size_t)M * D * 2);
  u16* Abf     = (u16*)wsalloc((size_t)M * D * 2);

  // weights -> bf16
  cvt_f32_bf16<<<(D * D / 4 + 255) / 256, 256, 0, stream>>>(Wq, Wq_bf, D * D);
  cvt_f32_bf16<<<(D * 2 * D / 4 + 255) / 256, 256, 0, stream>>>(Wkv, Wkv_bf, D * 2 * D);
  cvt_f32_bf16<<<(D * D / 4 + 255) / 256, 256, 0, stream>>>(Wout, Wout_bf, D * D);

  // Q = x @ Wq  -> bf16 [B,H,N,hd]
  gemm_bf16_kernel<<<dim3(D / 128, M / 128), 256, 0, stream>>>(
      x, 0, Wq_bf, D, D, 0, Qbf, Qbf, D, nullptr, nullptr, N, H);
  // K,V = context @ Wkv -> bf16 [B,H,N,hd] each
  gemm_bf16_kernel<<<dim3(2 * D / 128, M / 128), 256, 0, stream>>>(
      ctx, 0, Wkv_bf, D, 2 * D, 0, Kbf, Vbf, D, nullptr, nullptr, N, H);

  // rotary (scale hd^-0.5 = 0.125 folded into Q)
  const int BH = B * H;
  const int pairs = BH * N * 32;
  rotary_kernel<<<(pairs + 255) / 256, 256, 0, stream>>>(Qbf, pos, N, 0.125f, pairs);
  rotary_kernel<<<(pairs + 255) / 256, 256, 0, stream>>>(Kbf, pos, N, 1.0f, pairs);

  // flash attention -> bf16 [B*N, D]
  flash_attn_kernel<<<dim3(N / 128, BH), 256, 0, stream>>>(Qbf, Kbf, Vbf, Abf, B, H, N);

  // out = attn_out @ Wout + b_out  (f32)
  gemm_bf16_kernel<<<dim3(D / 128, M / 128), 256, 0, stream>>>(
      Abf, 1, Wout_bf, D, D, 1, nullptr, nullptr, 0, out, bout, N, H);
}
